// CAFFE_76914274337104
// MI455X (gfx1250) — compile-verified, matrix operand order checked by reading, not run on hardware
//
#include <hip/hip_runtime.h>

// CDNA5 / gfx1250, wave32. Matrix path: v_wmma_f32_16x16x32_f16 fed from LDS
// via conflict-free ds_load_b128 (B fragment = 2 x 16B reads, zero shuffles).
typedef _Float16 v16h __attribute__((ext_vector_type(16)));
typedef float    v8f  __attribute__((ext_vector_type(8)));
typedef unsigned v8u  __attribute__((ext_vector_type(8)));

__device__ __forceinline__ unsigned pk_and(float a, float b, unsigned m) {
  return __builtin_bit_cast(unsigned, __builtin_amdgcn_cvt_pkrtz(a, b)) & m;
}

// Local correlation (banded GEMM) + softmax + soft-argmax flow + confidence.
// Block = 128 threads = 4 waves = 4 consecutive rows of one 16-pixel tile.
// Shared ref slab (rows y0-PAD..y0+3+PAD, 32 cols, 32 channels) staged in LDS
// as pad-masked f16 pairs, double buffered: ~2.7x less L2 traffic.
// Slab layout [row][col][chan-pair]: pair stride 1, col stride 20 words
// (16B-aligned, 20 = 4*5 with 5 coprime to 16 => b128 reads tile all 64 banks
// exactly twice = minimum). aff overlays slab0 after the k-loop.
template<int C, int H, int W, int WS>
__global__ __launch_bounds__(128) void corr_flow_kernel(
    const float* __restrict__ tgt, const float* __restrict__ ref,
    float* __restrict__ flow_out, float* __restrict__ conf_out)
{
  constexpr int K    = WS * WS;
  constexpr int PAD  = WS / 2;
  constexpr int TX   = (W + 15) / 16;
  constexpr int KST  = C / 32;             // 32-channel k-steps
  constexpr int HW   = H * W;
  constexpr int G    = 4;                  // rows per block
  constexpr int R    = G + WS - 1;         // staged ref rows
  constexpr int LCS  = 20;                 // col stride (words): 16B-aligned, bank-safe
  constexpr int LCN  = 36;                 // col slots (32 staged + junk for WS<9)
  constexpr int RS   = LCN * LCS;          // row stride (words) = 720
  constexpr int SLAB = R * RS;             // words per buffer

  extern __shared__ unsigned smem[];       // [2*SLAB]; aff overlays buffer 0 later

  const int tid  = threadIdx.x;
  const int w    = tid >> 5;               // wave id = row offset
  const int lane = tid & 31;
  const int l15  = lane & 15;
  int t  = blockIdx.x;
  const int tx = t % TX;        t /= TX;
  const int yg = t % (H / G);
  const int b  = t / (H / G);
  const int y0 = yg * G;
  const int y  = y0 + w;                   // this wave's output row
  int x0 = tx * 16; if (x0 > W - 16) x0 = W - 16;   // edge tiles overlap
  const int xw = x0 - 4;                   // staged window origin (mult of 4)

  const int rb    = b * C * HW;
  const int half  = (lane >= 16) ? 8 : 0;  // f16 A/B K-packing: hi half owns K+8
  const int abase = __builtin_amdgcn_readfirstlane(rb + y * W + x0);

  // Fill roles: thread covers (chan-pair fpv = tid&15, col quad flc4).
  const int fpv  = tid & 15;
  const int flc4 = (tid >> 4) * 4;

  // Zero the junk col slots (lc 32..35) once; fills never touch them.
  for (int i = tid; i < 2 * R * 4 * 16; i += 128) {
    const int bfi = i / (R * 4 * 16);
    const int r   = i % (R * 4 * 16);
    smem[bfi * SLAB + (r / 64) * RS + (32 + (r % 64) / 16) * LCS + (r % 16)] = 0u;
  }

  v8f acc[WS][2];
  #pragma unroll
  for (int dy = 0; dy < WS; ++dy)
    #pragma unroll
    for (int nt = 0; nt < 2; ++nt)
      acc[dy][nt] = v8f{0.f,0.f,0.f,0.f,0.f,0.f,0.f,0.f};

  // ---- staged slab fill: masked+converted f16 pairs --------------------
  auto fill = [&](int bf, int ks) {
    const int coff = ks * 32 * HW;
    unsigned* dst0 = smem + bf * SLAB;
    #pragma unroll
    for (int lr = 0; lr < R; ++lr) {
      const int  ry  = y0 + lr - PAD;
      const bool okr = (ry >= 0) && (ry < H);
      const int  ryc = min(max(ry, 0), H - 1);
      const float* p0 = ref + (unsigned)(rb + ryc * W + 2 * fpv * HW + coff);
      const float* p1 = p0 + HW;
      const int  cb = xw + flc4;
      unsigned* dp = dst0 + lr * RS + flc4 * LCS + fpv;   // +j*LCS per col
      if (okr && cb >= 0 && cb + 4 <= W) {          // aligned interior quad
        const float4 r0 = *reinterpret_cast<const float4*>(p0 + cb);
        const float4 r1 = *reinterpret_cast<const float4*>(p1 + cb);
        dp[0 * LCS] = pk_and(r0.x, r1.x, 0xFFFFFFFFu);
        dp[1 * LCS] = pk_and(r0.y, r1.y, 0xFFFFFFFFu);
        dp[2 * LCS] = pk_and(r0.z, r1.z, 0xFFFFFFFFu);
        dp[3 * LCS] = pk_and(r0.w, r1.w, 0xFFFFFFFFu);
      } else {                                      // clamped + masked edge
        #pragma unroll
        for (int j = 0; j < 4; ++j) {
          const int  col = cb + j;
          const bool ok  = okr && (col >= 0) && (col < W);
          const int  cc  = min(max(col, 0), W - 1);
          dp[j * LCS] = pk_and(p0[cc], p1[cc], ok ? 0xFFFFFFFFu : 0u);
        }
      }
    }
  };

  // ---- GEMM from LDS ---------------------------------------------------
  auto compute = [&](int bf, int ks) {
    const int coff = ks * 32 * HW;
    // A fragment: direct global NT loads (tgt is single-use).
    const float* tp = tgt + (unsigned)(abase + coff);
    v8u au;
    #pragma unroll
    for (int v = 0; v < 8; ++v) {
      const int kb = ((v < 4) ? 2 * v : 16 + 2 * (v - 4)) + half;
      const float r0 = __builtin_nontemporal_load(&tp[l15 + kb * HW]);
      const float r1 = __builtin_nontemporal_load(&tp[l15 + (kb + 1) * HW]);
      au[v] = __builtin_bit_cast(unsigned, __builtin_amdgcn_cvt_pkrtz(r0, r1));
    }
    const v16h a = __builtin_bit_cast(v16h, au);

    const unsigned* sbp = smem + bf * SLAB + ((lane >= 16) ? 4 : 0);
    #pragma unroll
    for (int dy = 0; dy < WS; ++dy) {
      #pragma unroll
      for (int nt = 0; nt < 2; ++nt) {
        // B fragment: two aligned 16B LDS reads -> bu[0..3], bu[4..7].
        const unsigned* q =
            sbp + (w + dy) * RS + (nt * 16 + l15 + (4 - PAD)) * LCS;
        const uint4 lo = *reinterpret_cast<const uint4*>(q);      // pv 0..3 (+4 hi)
        const uint4 hi = *reinterpret_cast<const uint4*>(q + 8);  // pv 8..11 (+4 hi)
        const v8u bu = {lo.x, lo.y, lo.z, lo.w, hi.x, hi.y, hi.z, hi.w};
        const v16h bb = __builtin_bit_cast(v16h, bu);
        acc[dy][nt] = __builtin_amdgcn_wmma_f32_16x16x32_f16(
            false, a, false, bb, (short)0, acc[dy][nt], false, false);
      }
    }
  };

  fill(0, 0);
  __syncthreads();
  for (int ks = 0; ks < KST; ++ks) {
    const int bf = ks & 1;
    if (ks + 1 < KST) fill(bf ^ 1, ks + 1);   // prefetch next slab under WMMAs
    compute(bf, ks);
    __syncthreads();                          // readers done + next slab ready
  }

  // Scatter useful diagonal band to LDS (overlays dead slab0):
  // aff[x*K + dy*WS + dx']; C/D layout: vgpr v, lane l -> M=v+8*(l>=16), N=l&15.
  float* aff = (float*)smem + w * 16 * K;
  #pragma unroll
  for (int dy = 0; dy < WS; ++dy)
    #pragma unroll
    for (int nt = 0; nt < 2; ++nt)
      #pragma unroll
      for (int v = 0; v < 8; ++v) {
        const int x   = v + ((lane >= 16) ? 8 : 0);
        const int u   = l15 + nt * 16;
        const int dxp = u - x;
        if (dxp >= 0 && dxp < WS)
          aff[x * K + dy * WS + dxp] = acc[dy][nt][v];
      }
  __syncthreads();

  // Softmax + soft-argmax + confidence: one lane per pixel (per wave/row).
  if (lane < 16) {
    const float* av = &aff[lane * K];
    float m = av[0];
    for (int k = 1; k < K; ++k) m = fmaxf(m, av[k]);
    float Z = 0.f, Sy = 0.f, Sx = 0.f;
    for (int k = 0; k < K; ++k) {
      const float e = __expf(av[k] - m);
      Z  += e;
      Sy += e * (float)(k / WS - PAD);
      Sx += e * (float)(k % WS - PAD);
    }
    const float inv = 1.0f / Z;          // max(softmax) = 1/Z (argmax term)
    const size_t pix = ((size_t)b * H + y) * W + (x0 + lane);
    flow_out[pix * 2 + 0] = Sy * inv;    // d=0: ii (dy) weights
    flow_out[pix * 2 + 1] = Sx * inv;    // d=1: jj (dx) weights
    conf_out[pix]         = m * inv;     // max(aff) * max(sm)
  }
}

static constexpr int smem_bytes(int WS) {
  const int R = 4 + WS - 1;
  return 2 * (R * 36 * 20) * 4;          // aff overlays buffer 0 (always smaller)
}

extern "C" void kernel_launch(void* const* d_in, const int* in_sizes, int n_in,
                              void* d_out, int out_size, void* d_ws, size_t ws_size,
                              hipStream_t stream) {
  (void)in_sizes; (void)n_in; (void)out_size; (void)d_ws; (void)ws_size;
  // setup_inputs() dict order: tgt0, ref0, tgt1, ref1, tgt2, ref2
  const float* tgt0 = (const float*)d_in[0];
  const float* ref0 = (const float*)d_in[1];
  const float* tgt1 = (const float*)d_in[2];
  const float* ref1 = (const float*)d_in[3];
  const float* tgt2 = (const float*)d_in[4];
  const float* ref2 = (const float*)d_in[5];

  float* out   = (float*)d_out;
  // return order: flow0, flow1, flow2, conf0, conf1, conf2 (flat concat)
  float* flow0 = out;                 // 8*96*160*2 = 245760
  float* flow1 = out + 245760;        // 8*48*80*2  =  61440
  float* flow2 = out + 307200;        // 8*24*40*2  =  15360
  float* conf0 = out + 322560;        // 8*96*160   = 122880
  float* conf1 = out + 445440;        // 8*48*80    =  30720
  float* conf2 = out + 476160;        // 8*24*40    =   7680

  corr_flow_kernel<256,  96, 160, 9>
      <<<8 * (96/4) * 10, 128, smem_bytes(9), stream>>>(tgt0, ref0, flow0, conf0);
  corr_flow_kernel<512,  48,  80, 5>
      <<<8 * (48/4) *  5, 128, smem_bytes(5), stream>>>(tgt1, ref1, flow1, conf1);
  corr_flow_kernel<1024, 24,  40, 3>
      <<<8 * (24/4) *  3, 128, smem_bytes(3), stream>>>(tgt2, ref2, flow2, conf2);
}
